// MambaNet_37134287241962
// MI455X (gfx1250) — matrix operand order repaired and checked by reading, not compiled, not run on hardware
//
#include <hip/hip_runtime.h>

// ---------------------------------------------------------------------------
// MambaNet forward for gfx1250 (MI455X, wave32, WMMA).
// GEMMs use v_wmma_f32_16x16x32_bf16 (f32 accum) with 4x M register blocking
// per wave (B fragment reused across 4 WMMAs per K-step). Selective scan uses
// a 3-pass chunked parallel scan (96 chunks x 96 steps). Precision-critical
// scalar ops (softplus/exp/SiLU/GELU/LN/BN) remain f32.
// ---------------------------------------------------------------------------

typedef __attribute__((ext_vector_type(16))) __bf16 v16bf;
typedef __attribute__((ext_vector_type(8)))  float  v8f;

#define BATCH  2
#define HW     96
#define LSEQ   9216            // 96*96
#define MTOK   18432           // BATCH*LSEQ
#define DM     64
#define DI     128
#define NS     16
#define NB     6
#define CHUNKS 96
#define CSZ    96              // CHUNKS*CSZ == LSEQ
#define WSTRIDE 63488          // bf16 weight arena stride per block

__device__ __forceinline__ float sigf(float x) { return 1.f / (1.f + __expf(-x)); }

// ------------------------- weight prep (f32 -> bf16 arena, A = -exp(A_log))
__global__ void prep_weights_k(const float* ipw, const float* xpw, const float* opw,
                               const float* m1w, const float* m2w, const float* Alog,
                               __bf16* wb, float* Aneg, int n) {
  int g = blockIdx.x * blockDim.x + threadIdx.x;
  if (g >= n) return;
  const int n0 = NB * 256 * 64;     // in_proj
  const int n1 = NB * 48 * 128;     // x_proj (padded 36->48)
  const int n2 = NB * 64 * 128;     // out_proj
  const int n3 = NB * 256 * 64;     // mlp_w1
  const int n4 = NB * 64 * 256;     // mlp_w2
  if (g < n0) {
    int i = g / 16384, loc = g % 16384;
    wb[(size_t)i * WSTRIDE + loc] = (__bf16)ipw[g];
  } else if (g < n0 + n1) {
    int g1 = g - n0;
    int i = g1 / 6144, loc = g1 % 6144;
    int r = loc / 128, k = loc % 128;
    float v = (r < 36) ? xpw[((size_t)i * 36 + r) * 128 + k] : 0.f;
    wb[(size_t)i * WSTRIDE + 16384 + loc] = (__bf16)v;
  } else if (g < n0 + n1 + n2) {
    int g2 = g - n0 - n1;
    int i = g2 / 8192, loc = g2 % 8192;
    wb[(size_t)i * WSTRIDE + 22528 + loc] = (__bf16)opw[g2];
  } else if (g < n0 + n1 + n2 + n3) {
    int g3 = g - n0 - n1 - n2;
    int i = g3 / 16384, loc = g3 % 16384;
    wb[(size_t)i * WSTRIDE + 30720 + loc] = (__bf16)m1w[g3];
  } else if (g < n0 + n1 + n2 + n3 + n4) {
    int g4 = g - n0 - n1 - n2 - n3;
    int i = g4 / 16384, loc = g4 % 16384;
    wb[(size_t)i * WSTRIDE + 47104 + loc] = (__bf16)m2w[g4];
  } else {
    int g5 = g - n0 - n1 - n2 - n3 - n4;   // NB*128*16
    Aneg[g5] = -__expf(Alog[g5]);
  }
}

// ------------------------- generic wave32 WMMA GEMM: C(MxN) = A(MxK) * B^T(NxK)
// A,B bf16 row-major. Each wave computes a 64x16 output strip (4 M-subtiles),
// reusing the B fragment across 4 WMMAs per K-step. 4 waves/block span N tiles.
// mode 0: store f32   mode 1: +bias, exact GELU, store bf16   mode 2: (+bias)+res, store f32
__global__ void gemm_wmma_k(const __bf16* __restrict__ A, const __bf16* __restrict__ B,
                            float* outF, __bf16* outH,
                            const float* bias, const float* res,
                            int N, int K, int lda, int ldb, int ldc, int mode) {
  const int wave = threadIdx.x >> 5;
  const int lane = threadIdx.x & 31;
  const int Ntiles = (N + 15) >> 4;
  const int nt = blockIdx.y * 4 + wave;
  if (nt >= Ntiles) return;
  const int m0 = blockIdx.x << 6;            // 64-row strip
  const int n0 = nt << 4;
  const int row = lane & 15;
  const int hi  = (lane >> 4) & 1;
  const __bf16* __restrict__ Arow = A + (size_t)(m0 + row) * lda;
  const __bf16* __restrict__ Brow = B + (size_t)(n0 + row) * ldb;
  const size_t astep = (size_t)16 * lda;     // next M-subtile
  v8f acc[4];
#pragma unroll
  for (int t = 0; t < 4; ++t) acc[t] = (v8f){};
  for (int k0 = 0; k0 < K; k0 += 32) {
    // B 32x16 fragment: lane = col N; lanes 0-15 hold K=k0..k0+15,
    // lanes 16-31 hold K=k0+16..k0+31 (contiguous k order, matches A). Loaded ONCE per K-step.
    v16bf bv;
    const int kb = k0 + hi * 16;
#pragma unroll
    for (int i = 0; i < 16; ++i) bv[i] = Brow[kb + i];
    // A 16-bit 16x32 fragment per M-subtile: lanes 0-15 rows M,
    // K=k0..k0+15; lanes 16-31 same rows, K=k0+16..k0+31 (matches B).
    const int ka = k0 + hi * 16;
#pragma unroll
    for (int t = 0; t < 4; ++t) {
      const __bf16* __restrict__ ap = Arow + (size_t)t * astep;
      v16bf av;
#pragma unroll
      for (int i = 0; i < 8; ++i) {
        av[i]     = ap[ka + i];
        av[i + 8] = ap[ka + 8 + i];
      }
      acc[t] = __builtin_amdgcn_wmma_f32_16x16x32_bf16(false, av, false, bv,
                                                       (short)0, acc[t], false, false);
    }
  }
  // C/D layout: element (m = r + hi*8, n = lane&15) in acc[t][r]
  const int ocol = n0 + row;
#pragma unroll
  for (int t = 0; t < 4; ++t) {
    size_t obase = (size_t)(m0 + t * 16 + hi * 8) * ldc + ocol;
    if (mode == 0) {
#pragma unroll
      for (int r = 0; r < 8; ++r) outF[obase + (size_t)r * ldc] = acc[t][r];
    } else if (mode == 1) {
      const float bb = bias[ocol];
#pragma unroll
      for (int r = 0; r < 8; ++r) {
        float v = acc[t][r] + bb;
        float g = 0.5f * v * (1.f + erff(v * 0.70710678118654752f));
        outH[obase + (size_t)r * ldc] = (__bf16)g;
      }
    } else {
      const float bb = bias ? bias[ocol] : 0.f;
#pragma unroll
      for (int r = 0; r < 8; ++r) {
        size_t idx = obase + (size_t)r * ldc;
        outF[idx] = res[idx] + acc[t][r] + bb;
      }
    }
  }
}

// ------------------------- LayerNorm over C=64 + bf16 cast
__global__ void ln_cast_k(const float* __restrict__ h, const float* g, const float* b,
                          __bf16* out, int n) {
  int t = blockIdx.x * blockDim.x + threadIdx.x;
  if (t >= n) return;
  const float* r = h + (size_t)t * 64;
  float mu = 0.f;
#pragma unroll
  for (int k = 0; k < 64; ++k) mu += r[k];
  mu *= (1.f / 64.f);
  float var = 0.f;
#pragma unroll
  for (int k = 0; k < 64; ++k) { float d = r[k] - mu; var += d * d; }
  var *= (1.f / 64.f);
  float inv = rsqrtf(var + 1e-5f);
  __bf16* o = out + (size_t)t * 64;
#pragma unroll
  for (int k = 0; k < 64; ++k) o[k] = (__bf16)((r[k] - mu) * inv * g[k] + b[k]);
}

// ------------------------- causal depthwise conv1d (K=4) + SiLU
__global__ void dwconv_silu_k(const float* __restrict__ xz, const float* cw, const float* cb,
                              float* xs, __bf16* xsbf, int n) {
  int i = blockIdx.x * blockDim.x + threadIdx.x;
  if (i >= n) return;
  int d = i & 127;
  int t = i >> 7;
  int b = t / LSEQ, l = t % LSEQ;
  float acc = cb[d];
#pragma unroll
  for (int j = 0; j < 4; ++j) {
    int ll = l - 3 + j;
    if (ll >= 0) acc += cw[d * 4 + j] * xz[((size_t)(b * LSEQ + ll)) * 256 + d];
  }
  float s = acc * sigf(acc);
  xs[i] = s;
  xsbf[i] = (__bf16)s;
}

// ------------------------- delta = softplus(dt @ dtw.T + dtb)
__global__ void delta_k(const float* __restrict__ dbl, const float* dtw, const float* dtb,
                        float* delta, int n) {
  int i = blockIdx.x * blockDim.x + threadIdx.x;
  if (i >= n) return;
  int d = i & 127;
  int t = i >> 7;
  const float* dr = dbl + (size_t)t * 48;
  float a = dtb[d] + dtw[d * 4 + 0] * dr[0] + dtw[d * 4 + 1] * dr[1]
                   + dtw[d * 4 + 2] * dr[2] + dtw[d * 4 + 3] * dr[3];
  delta[i] = (a > 20.f) ? a : log1pf(__expf(a));
}

// ------------------------- chunked selective scan
// pass1: per-chunk local scan from h0=0; record decay product P and end state.
__global__ void scan_pass1_k(const float* __restrict__ dbl, const float* __restrict__ delta,
                             const float* __restrict__ xs, const float* __restrict__ Aneg,
                             float* Pbuf, float* hEbuf) {
  int b = blockIdx.x / CHUNKS;
  int c = blockIdx.x % CHUNKS;
  int d = threadIdx.x;              // 128
  __shared__ float bc[32];
  float A[NS], h[NS], P[NS];
#pragma unroll
  for (int n = 0; n < NS; ++n) { A[n] = Aneg[d * NS + n]; h[n] = 0.f; P[n] = 1.f; }
  for (int s = 0; s < CSZ; ++s) {
    int tok = b * LSEQ + c * CSZ + s;
    if (d < 32) bc[d] = dbl[(size_t)tok * 48 + 4 + d];
    __syncthreads();
    float dv = delta[(size_t)tok * 128 + d];
    float xv = xs[(size_t)tok * 128 + d];
    float dx = dv * xv;
#pragma unroll
    for (int n = 0; n < NS; ++n) {
      float a = __expf(dv * A[n]);
      h[n] = a * h[n] + dx * bc[n];
      P[n] *= a;
    }
    __syncthreads();
  }
  size_t base = ((size_t)(b * 128 + d) * CHUNKS + c) * NS;
#pragma unroll
  for (int n = 0; n < NS; ++n) { Pbuf[base + n] = P[n]; hEbuf[base + n] = h[n]; }
}

// pass2: tiny sequential carry across chunks (one block).
__global__ void scan_pass2_k(const float* __restrict__ Pbuf, const float* __restrict__ hEbuf,
                             float* initbuf) {
  int t = threadIdx.x;              // 256 = (b*128+d)
  float carry[NS];
#pragma unroll
  for (int n = 0; n < NS; ++n) carry[n] = 0.f;
  size_t base = (size_t)t * CHUNKS * NS;
  for (int c = 0; c < CHUNKS; ++c) {
    size_t o = base + (size_t)c * NS;
#pragma unroll
    for (int n = 0; n < NS; ++n) {
      initbuf[o + n] = carry[n];
      carry[n] = Pbuf[o + n] * carry[n] + hEbuf[o + n];
    }
  }
}

// pass3: replay chunks with correct init state; fuse D-skip + SiLU(z) gate + bf16 cast.
__global__ void scan_pass3_k(const float* __restrict__ dbl, const float* __restrict__ delta,
                             const float* __restrict__ xs, const float* __restrict__ xz,
                             const float* __restrict__ Aneg, const float* __restrict__ Dp,
                             const float* __restrict__ initbuf, __bf16* ybf) {
  int b = blockIdx.x / CHUNKS;
  int c = blockIdx.x % CHUNKS;
  int d = threadIdx.x;
  __shared__ float bc[32];
  float A[NS], h[NS];
  size_t ibase = ((size_t)(b * 128 + d) * CHUNKS + c) * NS;
#pragma unroll
  for (int n = 0; n < NS; ++n) { A[n] = Aneg[d * NS + n]; h[n] = initbuf[ibase + n]; }
  float Dv = Dp[d];
  for (int s = 0; s < CSZ; ++s) {
    int tok = b * LSEQ + c * CSZ + s;
    if (d < 32) bc[d] = dbl[(size_t)tok * 48 + 4 + d];
    __syncthreads();
    float dv = delta[(size_t)tok * 128 + d];
    float xv = xs[(size_t)tok * 128 + d];
    float dx = dv * xv;
    float y = 0.f;
#pragma unroll
    for (int n = 0; n < NS; ++n) {
      float a = __expf(dv * A[n]);
      h[n] = a * h[n] + dx * bc[n];
      y += h[n] * bc[16 + n];
    }
    y += xv * Dv;
    float z = xz[(size_t)tok * 256 + 128 + d];
    y *= z * sigf(z);
    ybf[(size_t)tok * 128 + d] = (__bf16)y;
    __syncthreads();
  }
}

// ------------------------- stem
__global__ void stem_conv1_k(const float* __restrict__ x, const float* w1, const float* scw,
                             float* t1, float* tS, int n) {
  int i = blockIdx.x * blockDim.x + threadIdx.x;
  if (i >= n) return;
  int co = i & 63;
  int p = i >> 6;
  int b = p / LSEQ;
  int yy = (p % LSEQ) / HW, xx = p % HW;
  float acc = 0.f;
#pragma unroll
  for (int dy = 0; dy < 3; ++dy)
#pragma unroll
    for (int dx = 0; dx < 3; ++dx) {
      int iy = yy + dy - 1, ix = xx + dx - 1;
      if (iy >= 0 && iy < HW && ix >= 0 && ix < HW)
        acc += w1[co * 9 + dy * 3 + dx] * x[(size_t)b * LSEQ + iy * HW + ix];
    }
  t1[i] = acc;
  tS[i] = scw[co] * x[(size_t)b * LSEQ + yy * HW + xx];
}

__global__ void bn_stats_k(const float* __restrict__ src, float* mean, float* var) {
  int c = blockIdx.x;                        // 64 channels
  __shared__ float s1[256], s2[256];
  float a = 0.f, a2 = 0.f;
  for (int i = threadIdx.x; i < MTOK; i += 256) {
    float v = src[(size_t)i * 64 + c];
    a += v; a2 += v * v;
  }
  s1[threadIdx.x] = a; s2[threadIdx.x] = a2;
  __syncthreads();
  for (int s = 128; s > 0; s >>= 1) {
    if (threadIdx.x < s) { s1[threadIdx.x] += s1[threadIdx.x + s]; s2[threadIdx.x] += s2[threadIdx.x + s]; }
    __syncthreads();
  }
  if (threadIdx.x == 0) {
    float mu = s1[0] / (float)MTOK;
    mean[c] = mu;
    var[c] = s2[0] / (float)MTOK - mu * mu;
  }
}

__global__ void stem_bn1_k(const float* t1, const float* tS,
                           const float* mu1, const float* v1, const float* g1, const float* b1,
                           const float* muS, const float* vS, const float* gS, const float* bS,
                           float* h1, float* res, int n) {
  int i = blockIdx.x * blockDim.x + threadIdx.x;
  if (i >= n) return;
  int c = i & 63;
  float a = (t1[i] - mu1[c]) * rsqrtf(v1[c] + 1e-5f) * g1[c] + b1[c];
  h1[i] = a > 0.f ? a : 0.f;
  res[i] = (tS[i] - muS[c]) * rsqrtf(vS[c] + 1e-5f) * gS[c] + bS[c];
}

__global__ void stem_conv2_k(const float* __restrict__ h1, const float* __restrict__ w2,
                             float* t2, int n) {
  int i = blockIdx.x * blockDim.x + threadIdx.x;
  if (i >= n) return;
  int co = i & 63;
  int p = i >> 6;
  int b = p / LSEQ;
  int yy = (p % LSEQ) / HW, xx = p % HW;
  float acc = 0.f;
  for (int dy = 0; dy < 3; ++dy) {
    int iy = yy + dy - 1;
    if (iy < 0 || iy >= HW) continue;
    for (int dx = 0; dx < 3; ++dx) {
      int ix = xx + dx - 1;
      if (ix < 0 || ix >= HW) continue;
      const float* hp = h1 + ((size_t)b * LSEQ + iy * HW + ix) * 64;
      const float* wp = w2 + ((size_t)co * 64) * 9 + dy * 3 + dx;
#pragma unroll 8
      for (int ci = 0; ci < 64; ++ci) acc += hp[ci] * wp[(size_t)ci * 9];
    }
  }
  t2[i] = acc;
}

__global__ void stem_bn2_add_k(const float* t2, const float* res,
                               const float* mu2, const float* v2, const float* g2, const float* b2,
                               float* h, int n) {
  int i = blockIdx.x * blockDim.x + threadIdx.x;
  if (i >= n) return;
  int c = i & 63;
  float a = (t2[i] - mu2[c]) * rsqrtf(v2[c] + 1e-5f) * g2[c] + b2[c] + res[i];
  h[i] = a > 0.f ? a : 0.f;
}

// ------------------------- pooling + fc
__global__ void pool_k(const float* __restrict__ h, float* pooled) {
  int bc = blockIdx.x;                 // 0..127: b = bc>>6, c = bc&63
  int b = bc >> 6, c = bc & 63;
  __shared__ float s[256];
  float a = 0.f;
  for (int i = threadIdx.x; i < LSEQ; i += 256)
    a += h[((size_t)b * LSEQ + i) * 64 + c];
  s[threadIdx.x] = a;
  __syncthreads();
  for (int st = 128; st > 0; st >>= 1) {
    if (threadIdx.x < st) s[threadIdx.x] += s[threadIdx.x + st];
    __syncthreads();
  }
  if (threadIdx.x == 0) pooled[bc] = s[0] / (float)LSEQ;
}

__global__ void fc_k(const float* pooled, const float* fw, const float* fb, float* out) {
  __shared__ float s0[64], s1[64];
  int t = threadIdx.x;                 // 64
  s0[t] = pooled[t] * fw[t];
  s1[t] = pooled[64 + t] * fw[t];
  __syncthreads();
  for (int st = 32; st > 0; st >>= 1) {
    if (t < st) { s0[t] += s0[t + st]; s1[t] += s1[t + st]; }
    __syncthreads();
  }
  if (t == 0) { out[0] = s0[0] + fb[0]; out[1] = s1[0] + fb[0]; }
}

// ---------------------------------------------------------------------------
extern "C" void kernel_launch(void* const* d_in, const int* in_sizes, int n_in,
                              void* d_out, int out_size, void* d_ws, size_t ws_size,
                              hipStream_t stream) {
  (void)in_sizes; (void)n_in; (void)out_size; (void)ws_size;
  const float* x      = (const float*)d_in[0];
  const float* conv1w = (const float*)d_in[1];
  const float* bn1g   = (const float*)d_in[2];
  const float* bn1b   = (const float*)d_in[3];
  const float* conv2w = (const float*)d_in[4];
  const float* bn2g   = (const float*)d_in[5];
  const float* bn2b   = (const float*)d_in[6];
  const float* scw    = (const float*)d_in[7];
  const float* scg    = (const float*)d_in[8];
  const float* scb    = (const float*)d_in[9];
  const float* ln1g   = (const float*)d_in[10];
  const float* ln1b   = (const float*)d_in[11];
  const float* ipw    = (const float*)d_in[12];
  const float* cw     = (const float*)d_in[13];
  const float* cb     = (const float*)d_in[14];
  const float* xpw    = (const float*)d_in[15];
  const float* dtw    = (const float*)d_in[16];
  const float* dtb    = (const float*)d_in[17];
  const float* Alog   = (const float*)d_in[18];
  const float* Dpar   = (const float*)d_in[19];
  const float* opw    = (const float*)d_in[20];
  const float* ln2g   = (const float*)d_in[21];
  const float* ln2b   = (const float*)d_in[22];
  const float* m1w    = (const float*)d_in[23];
  const float* m1b    = (const float*)d_in[24];
  const float* m2w    = (const float*)d_in[25];
  const float* m2b    = (const float*)d_in[26];
  const float* fcw    = (const float*)d_in[27];
  const float* fcb    = (const float*)d_in[28];
  float* out = (float*)d_out;

  char* ws = (char*)d_ws;
  size_t off = 0;
  auto alloc = [&](size_t bytes) { size_t o = off; off = (off + bytes + 255) & ~(size_t)255; return o; };
  size_t o_h    = alloc((size_t)MTOK * 64 * 4);
  size_t o_xz   = alloc((size_t)MTOK * 256 * 4);
  size_t o_xs   = alloc((size_t)MTOK * 128 * 4);
  size_t o_dlt  = alloc((size_t)MTOK * 128 * 4);
  size_t o_dbl  = alloc((size_t)MTOK * 48 * 4);
  size_t o_ubf  = alloc((size_t)MTOK * 64 * 2);
  size_t o_gbf  = alloc((size_t)MTOK * 256 * 2);
  size_t o_xsbf = alloc((size_t)MTOK * 128 * 2);
  size_t o_ybf  = alloc((size_t)MTOK * 128 * 2);
  size_t o_wbf  = alloc((size_t)NB * WSTRIDE * 2);
  size_t o_An   = alloc((size_t)NB * 2048 * 4);
  size_t o_P    = alloc((size_t)2 * 128 * CHUNKS * NS * 4);
  size_t o_hE   = alloc((size_t)2 * 128 * CHUNKS * NS * 4);
  size_t o_ini  = alloc((size_t)2 * 128 * CHUNKS * NS * 4);
  size_t o_st   = alloc(64 * 6 * 4);
  size_t o_pool = alloc(128 * 4);

  float*  h     = (float*)(ws + o_h);
  float*  xz    = (float*)(ws + o_xz);
  float*  xs    = (float*)(ws + o_xs);
  float*  dlt   = (float*)(ws + o_dlt);
  float*  dbl   = (float*)(ws + o_dbl);
  __bf16* ubf   = (__bf16*)(ws + o_ubf);
  __bf16* gbf   = (__bf16*)(ws + o_gbf);
  __bf16* xsbf  = (__bf16*)(ws + o_xsbf);
  __bf16* ybf   = (__bf16*)(ws + o_ybf);
  __bf16* wbf   = (__bf16*)(ws + o_wbf);
  float*  Aneg  = (float*)(ws + o_An);
  float*  Pbuf  = (float*)(ws + o_P);
  float*  hEbuf = (float*)(ws + o_hE);
  float*  inib  = (float*)(ws + o_ini);
  float*  st    = (float*)(ws + o_st);
  float*  pooled= (float*)(ws + o_pool);
  // stem temps alias xz (needs 3 * MTOK*64*4 <= MTOK*256*4) and xs
  float* t1  = xz;
  float* tS  = xz + (size_t)MTOK * 64;
  float* t2  = xz + (size_t)MTOK * 128;
  float* res = xs;
  float *mu1 = st, *v1 = st + 64, *muS = st + 128, *vS = st + 192, *mu2 = st + 256, *v2 = st + 320;

  const int EL = MTOK * 64;        // 1,179,648
  const int ED = MTOK * 128;       // 2,359,296
  const int PREP_N = NB * (256 * 64 + 48 * 128 + 64 * 128 + 256 * 64 + 64 * 256) + NB * 2048;

  prep_weights_k<<<(PREP_N + 255) / 256, 256, 0, stream>>>(ipw, xpw, opw, m1w, m2w, Alog,
                                                           wbf, Aneg, PREP_N);
  // ---- stem
  stem_conv1_k<<<(EL + 255) / 256, 256, 0, stream>>>(x, conv1w, scw, t1, tS, EL);
  bn_stats_k<<<64, 256, 0, stream>>>(t1, mu1, v1);
  bn_stats_k<<<64, 256, 0, stream>>>(tS, muS, vS);
  stem_bn1_k<<<(EL + 255) / 256, 256, 0, stream>>>(t1, tS, mu1, v1, bn1g, bn1b,
                                                   muS, vS, scg, scb, h, res, EL);
  // h temporarily holds relu(bn1(conv1)); conv2 reads it, result to t2
  stem_conv2_k<<<(EL + 255) / 256, 256, 0, stream>>>(h, conv2w, t2, EL);
  bn_stats_k<<<64, 256, 0, stream>>>(t2, mu2, v2);
  stem_bn2_add_k<<<(EL + 255) / 256, 256, 0, stream>>>(t2, res, mu2, v2, bn2g, bn2b, h, EL);

  // ---- mamba blocks
  for (int i = 0; i < NB; ++i) {
    const __bf16* w_ip = wbf + (size_t)i * WSTRIDE + 0;
    const __bf16* w_xp = wbf + (size_t)i * WSTRIDE + 16384;
    const __bf16* w_op = wbf + (size_t)i * WSTRIDE + 22528;
    const __bf16* w_m1 = wbf + (size_t)i * WSTRIDE + 30720;
    const __bf16* w_m2 = wbf + (size_t)i * WSTRIDE + 47104;
    const float* An_i = Aneg + (size_t)i * 2048;

    ln_cast_k<<<(MTOK + 255) / 256, 256, 0, stream>>>(h, ln1g + i * 64, ln1b + i * 64, ubf, MTOK);
    // in_proj: (M,64)x(256,64)^T -> xz (M,256)
    gemm_wmma_k<<<dim3(MTOK / 64, 4), 128, 0, stream>>>(ubf, w_ip, xz, nullptr, nullptr, nullptr,
                                                        256, 64, 64, 64, 256, 0);
    dwconv_silu_k<<<(ED + 255) / 256, 256, 0, stream>>>(xz, cw + i * 512, cb + i * 128,
                                                        xs, xsbf, ED);
    // x_proj: (M,128)x(48,128)^T -> dbl (M,48) (cols >=36 are zero)
    gemm_wmma_k<<<dim3(MTOK / 64, 1), 128, 0, stream>>>(xsbf, w_xp, dbl, nullptr, nullptr, nullptr,
                                                        48, 128, 128, 128, 48, 0);
    delta_k<<<(ED + 255) / 256, 256, 0, stream>>>(dbl, dtw + i * 512, dtb + i * 128, dlt, ED);
    // chunked selective scan
    scan_pass1_k<<<BATCH * CHUNKS, 128, 0, stream>>>(dbl, dlt, xs, An_i, Pbuf, hEbuf);
    scan_pass2_k<<<1, 256, 0, stream>>>(Pbuf, hEbuf, inib);
    scan_pass3_k<<<BATCH * CHUNKS, 128, 0, stream>>>(dbl, dlt, xs, xz, An_i, Dpar + i * 128,
                                                     inib, ybf);
    // out_proj + residual: h += y @ ow.T
    gemm_wmma_k<<<dim3(MTOK / 64, 1), 128, 0, stream>>>(ybf, w_op, h, nullptr, nullptr, h,
                                                        64, 128, 128, 128, 64, 2);
    // MLP
    ln_cast_k<<<(MTOK + 255) / 256, 256, 0, stream>>>(h, ln2g + i * 64, ln2b + i * 64, ubf, MTOK);
    gemm_wmma_k<<<dim3(MTOK / 64, 4), 128, 0, stream>>>(ubf, w_m1, nullptr, gbf, m1b + i * 256,
                                                        nullptr, 256, 64, 64, 64, 256, 1);
    gemm_wmma_k<<<dim3(MTOK / 64, 1), 128, 0, stream>>>(gbf, w_m2, h, nullptr, m2b + i * 64, h,
                                                        64, 256, 256, 256, 64, 2);
  }

  // ---- head
  pool_k<<<128, 256, 0, stream>>>(h, pooled);
  fc_k<<<1, 64, 0, stream>>>(pooled, fcw, fcb, out);
}
